// OrthogonalPatchwise_91173565759920
// MI455X (gfx1250) — compile-verified, matrix operand order checked by reading, not run on hardware
//
#include <hip/hip_runtime.h>

typedef __attribute__((ext_vector_type(16))) _Float16 v16h;
typedef __attribute__((ext_vector_type(8)))  _Float16 h8;
typedef __attribute__((ext_vector_type(4)))  _Float16 h4;
typedef __attribute__((ext_vector_type(8)))  float    v8f;
typedef __attribute__((ext_vector_type(4)))  float    f4;

#define CH_STRIDE (224*224)
#define FEAT_K 1568   // 196 patches * 8 qubits
#define NCLS 1000

// Butterfly circuit wire pairs (12 pairs, repeated for 2 layers -> 24 gates)
__constant__ int c_w1[12] = {0,2,4,6,0,1,4,5,0,1,2,3};
__constant__ int c_w2[12] = {1,3,5,7,2,3,6,7,4,5,6,7};

__device__ __forceinline__ v8f wmma16x16x32(v16h a, v16h b, v8f c) {
  return __builtin_amdgcn_wmma_f32_16x16x32_f16(false, a, false, b, (short)0, c,
                                                false, false);
}

// ---------------- Kernel 0a: M = I (256x256, f32, row-major) ----------------
__global__ void k_init_identity(float* __restrict__ M) {
  int idx = blockIdx.x * 256 + threadIdx.x;          // 65536 elements
  M[idx] = ((idx >> 8) == (idx & 255)) ? 1.f : 0.f;
}

// ---------------- Kernel 0b: M <- G_g * M (one RBS gate) --------------------
__global__ void k_gate(float* __restrict__ M, const float* __restrict__ qp, int g) {
  int tid = blockIdx.x * 256 + threadIdx.x;          // 16384 = 64 groups * 256 cols
  int col = tid & 255;
  int grp = tid >> 8;                                // 0..63
  int pr  = g % 12;
  int b1  = 7 - c_w1[pr];                            // wire 0 = MSB
  int b2  = 7 - c_w2[pr];                            // b2 < b1
  float th = qp[g];
  float c = cosf(th), s = sinf(th);
  float c2 = c * c, s2 = s * s, cs = c * s;
  // deposit 6 free bits around positions b2 (low) and b1 (high)
  int m1 = (1 << b2) - 1;
  int t1 = ((grp & ~m1) << 1) | (grp & m1);
  int m2 = (1 << b1) - 1;
  int base = ((t1 & ~m2) << 1) | (t1 & m2);
  int i00 = base;
  int i01 = base | (1 << b2);
  int i10 = base | (1 << b1);
  int i11 = base | (1 << b1) | (1 << b2);
  float v00 = M[i00*256 + col];
  float v01 = M[i01*256 + col];
  float v10 = M[i10*256 + col];
  float v11 = M[i11*256 + col];
  M[i00*256 + col] =  c2*v00 + s2*v01 - cs*v10 + cs*v11;
  M[i01*256 + col] =  s2*v00 + c2*v01 + cs*v10 - cs*v11;
  M[i10*256 + col] =  cs*v00 - cs*v01 + c2*v10 + s2*v11;
  M[i11*256 + col] = -cs*v00 + cs*v01 + s2*v10 + c2*v11;
}

// ------- Kernel 0c: pack Wt = W^T as f16 in WMMA-B fragment order -----------
// Fragment (kt,nt): lane L holds n = nt*16 + (L&15); K values kt*32 + base..
// with base = (L>=16 ? 8 : 0), j<8 -> base+j, j>=8 -> base+8+j (K-interleave).
__global__ void k_pack(const float* __restrict__ M, _Float16* __restrict__ Bpack) {
  int idx  = blockIdx.x * 256 + threadIdx.x;         // 65536
  int j    = idx & 15;
  int lane = (idx >> 4) & 31;
  int frag = idx >> 9;                               // 0..127
  int nt   = frag & 15, kt = frag >> 4;
  int n  = nt * 16 + (lane & 15);
  int kb = kt * 32 + ((lane >> 4) << 3);
  int k  = kb + ((j < 8) ? j : (j + 8));
  // B[k][n] = W^T[k][n] = W[n][k]
  Bpack[idx] = (_Float16)M[n * 256 + k];
}

// ---- Main fused kernel: patches -> normalize -> U=V*W^T -> probs -> <Z_q> ---
__global__ __launch_bounds__(256)
void k_patch(const float* __restrict__ x, const _Float16* __restrict__ Bpack,
             float* __restrict__ feats) {
  __shared__ _Float16 Ah[64 * 264];   // 64 patch rows, padded stride (33 KB)
  __shared__ float    Pb[64 * 68];    // probs tile, quarter-N, padded (17 KB)
  __shared__ float    scale[64];

  int tid  = threadIdx.x;
  int lane = tid & 31;
  int wv   = tid >> 5;

  // ---- stage 64 patches: channel mean, f16, wide non-temporal streaming ----
  // 4 patches per iteration; each lane loads a 16-byte (float4) row chunk.
  {
    int i0  = tid >> 6;          // patch-within-group 0..3
    int e4  = tid & 63;          // 16B chunk id within patch (0..63)
    int rr2 = e4 >> 2;           // patch row 0..15
    int cc4 = (e4 & 3) << 2;     // patch col 0,4,8,12
    for (int ig = 0; ig < 16; ++ig) {
      int i  = ig * 4 + i0;
      int gp = blockIdx.x * 64 + i;
      int b = gp / 196, pp = gp % 196;
      int prr = pp / 14, pcc = pp % 14;
      size_t base = (size_t)b * (3 * CH_STRIDE)
                  + (size_t)(prr * 16 + rr2) * 224 + (pcc * 16 + cc4);
      f4 v0 = __builtin_nontemporal_load((const f4*)(x + base));
      f4 v1 = __builtin_nontemporal_load((const f4*)(x + base + CH_STRIDE));
      f4 v2 = __builtin_nontemporal_load((const f4*)(x + base + 2 * CH_STRIDE));
      f4 vm = (v0 + v1 + v2) * (1.f / 3.f);
      h4 hv;
      hv[0] = (_Float16)vm[0]; hv[1] = (_Float16)vm[1];
      hv[2] = (_Float16)vm[2]; hv[3] = (_Float16)vm[3];
      *(h4*)&Ah[i * 264 + (e4 << 2)] = hv;
    }
  }
  __syncthreads();

  // ---- deterministic per-patch norm (vectorized LDS reads) ----
  if (tid < 64) {
    float nn = 0.f;
    for (int e4i = 0; e4i < 64; ++e4i) {
      h4 hv = *(const h4*)&Ah[tid * 264 + (e4i << 2)];
      float a0 = (float)hv[0], a1 = (float)hv[1];
      float a2 = (float)hv[2], a3 = (float)hv[3];
      nn += a0 * a0 + a1 * a1 + a2 * a2 + a3 * a3;
    }
    scale[tid] = (nn > 0.f) ? rsqrtf(nn) : 1.f;
  }
  __syncthreads();
  {
    int i0 = tid >> 6;
    int e4 = tid & 63;
    for (int ig = 0; ig < 16; ++ig) {
      int i = ig * 4 + i0;
      float sc = scale[i];
      h4 hv = *(const h4*)&Ah[i * 264 + (e4 << 2)];
      h4 ho;
      ho[0] = (_Float16)((float)hv[0] * sc); ho[1] = (_Float16)((float)hv[1] * sc);
      ho[2] = (_Float16)((float)hv[2] * sc); ho[3] = (_Float16)((float)hv[3] * sc);
      *(h4*)&Ah[i * 264 + (e4 << 2)] = ho;
    }
  }

  // ---- GEMM 64x256x256 via WMMA, probs + signed Z reduction in 4 N-passes ----
  float zacc[2] = {0.f, 0.f};
  int mt   = wv & 3;                 // M-tile owned by this wave
  int ntl0 = (wv >> 2) * 2;          // 2 of 4 N-tiles per pass
  int am   = mt * 16 + (lane & 15);  // A-fragment row for this lane
  int koff = (lane >> 4) << 3;       // A/B K interleave offset

  for (int pass = 0; pass < 4; ++pass) {
    __syncthreads();                 // A ready / previous reduction done
    for (int f = 0; f < 2; ++f) {
      int ntp = ntl0 + f;            // 0..3 within pass
      int nt  = pass * 4 + ntp;      // global N-tile 0..15
      v8f acc = {};
#pragma unroll
      for (int kt = 0; kt < 8; ++kt) {
        int kb = kt * 32 + koff;
        const h8* ap = (const h8*)&Ah[am * 264 + kb];
        h8 alo = ap[0];
        h8 ahi = ap[2];              // +16 halfs
        v16h a = __builtin_shufflevector(alo, ahi,
                 0,1,2,3,4,5,6,7,8,9,10,11,12,13,14,15);
        v16h b = *(const v16h*)&Bpack[(((size_t)(kt * 16 + nt)) * 32 + lane) * 16];
        acc = wmma16x16x32(a, b, acc);
      }
      int nl    = ntp * 16 + (lane & 15);
      int mbase = mt * 16 + ((lane >> 4) << 3);   // D layout: VGPR r -> M = r (+8)
#pragma unroll
      for (int rr = 0; rr < 8; ++rr) {
        float u = acc[rr];
        Pb[(mbase + rr) * 68 + nl] = u * u;
      }
    }
    __syncthreads();
    // signed reduction: z[m][q] += sum_n sign(q,n) * probs[m][n]
#pragma unroll
    for (int u = 0; u < 2; ++u) {
      int o = tid + (u << 8);
      int m = o >> 3, q = o & 7, bp = 7 - q;
      float a2 = zacc[u];
      for (int nn2 = 0; nn2 < 64; ++nn2) {
        int gn = pass * 64 + nn2;
        float sg = ((gn >> bp) & 1) ? -1.f : 1.f;
        a2 = fmaf(sg, Pb[m * 68 + nn2], a2);
      }
      zacc[u] = a2;
    }
  }

#pragma unroll
  for (int u = 0; u < 2; ++u) {
    int o = tid + (u << 8);
    int m = o >> 3, q = o & 7;
    int gp = blockIdx.x * 64 + m;
    int b = gp / 196, pp = gp % 196;
    feats[(size_t)b * FEAT_K + pp * 8 + q] = zacc[u];
  }
}

// ---------------- FC: out(256x1000) = feats(256x1568) @ fc_w^T + b -----------
__global__ __launch_bounds__(256)
void k_fc(const float* __restrict__ feats, const float* __restrict__ fcw,
          const float* __restrict__ fcb, float* __restrict__ out) {
  int wv   = (blockIdx.x << 3) + (threadIdx.x >> 5);  // 0..1007
  int lane = threadIdx.x & 31;
  int mt = wv / 63, nt = wv % 63;
  int mrow = mt * 16 + (lane & 15);
  int n    = nt * 16 + (lane & 15);
  bool nvalid = n < NCLS;
  int nsafe = nvalid ? n : (NCLS - 1);   // clamped address, zeroed before convert
  int koff = (lane >> 4) << 3;

  v8f acc = {};
  for (int kt = 0; kt < 49; ++kt) {
    int kb = kt * 32 + koff;
    const float* arow = feats + (size_t)mrow * FEAT_K + kb;
    const float* brow = fcw + (size_t)nsafe * FEAT_K + kb;
    f4 a0 = *(const f4*)(arow);
    f4 a1 = *(const f4*)(arow + 4);
    f4 a2 = *(const f4*)(arow + 16);
    f4 a3 = *(const f4*)(arow + 20);
    f4 b0 = *(const f4*)(brow);
    f4 b1 = *(const f4*)(brow + 4);
    f4 b2 = *(const f4*)(brow + 16);
    f4 b3 = *(const f4*)(brow + 20);
    float bm = nvalid ? 1.f : 0.f;
    v16h a, b;
#pragma unroll
    for (int j = 0; j < 4; ++j) {
      a[j]      = (_Float16)a0[j];
      a[j + 4]  = (_Float16)a1[j];
      a[j + 8]  = (_Float16)a2[j];
      a[j + 12] = (_Float16)a3[j];
      b[j]      = (_Float16)(b0[j] * bm);
      b[j + 4]  = (_Float16)(b1[j] * bm);
      b[j + 8]  = (_Float16)(b2[j] * bm);
      b[j + 12] = (_Float16)(b3[j] * bm);
    }
    acc = wmma16x16x32(a, b, acc);
  }
  float bias = nvalid ? fcb[n] : 0.f;
#pragma unroll
  for (int rr = 0; rr < 8; ++rr) {
    int m = mt * 16 + rr + ((lane >> 4) << 3);
    if (nvalid) out[(size_t)m * NCLS + n] = acc[rr] + bias;
  }
}

extern "C" void kernel_launch(void* const* d_in, const int* in_sizes, int n_in,
                              void* d_out, int out_size, void* d_ws, size_t ws_size,
                              hipStream_t stream) {
  (void)in_sizes; (void)n_in; (void)out_size; (void)ws_size;
  const float* x   = (const float*)d_in[0];   // (256,3,224,224)
  const float* qp  = (const float*)d_in[1];   // (24,)
  const float* fcw = (const float*)d_in[2];   // (1000,1568)
  const float* fcb = (const float*)d_in[3];   // (1000,)
  float* out = (float*)d_out;                 // (256,1000)

  char* ws = (char*)d_ws;
  float*    M     = (float*)ws;               // 256 KB: circuit matrix W
  _Float16* Bpack = (_Float16*)(ws + 262144); // 128 KB: W^T packed B-fragments
  float*    feats = (float*)(ws + 393216);    // 1.57 MB: (256,1568)

  k_init_identity<<<256, 256, 0, stream>>>(M);
  for (int g = 0; g < 24; ++g)
    k_gate<<<64, 256, 0, stream>>>(M, qp, g);
  k_pack<<<256, 256, 0, stream>>>(M, Bpack);
  k_patch<<<784, 256, 0, stream>>>(x, Bpack, feats);   // 784*64 = 50176 patches
  k_fc<<<126, 256, 0, stream>>>(feats, fcw, fcb, out); // 1008 wave-tiles
}
